// AssistBranch_54778012893540
// MI455X (gfx1250) — matrix operand order repaired and checked by reading, not compiled, not run on hardware
//
#include <hip/hip_runtime.h>
#include <cstdint>
#include <cstddef>

// ---------------- problem constants ----------------
constexpr int Bc = 2, Tc = 2048, NCc = 64, Dc = 1024, Hc = 8, Lc = 2, MAXOFFc = 512;
constexpr int DHc = 128, DFFc = 4096, Rc = 1025, RPc = 1040;   // RP = R padded to /16
constexpr int Mtok = Bc * Tc;                                   // 4096 tokens
constexpr float SCALEc = 0.08838834764831845f;                  // 1/sqrt(128)

typedef __bf16 bf16_t;
typedef __attribute__((ext_vector_type(16))) __bf16 v16bf;
typedef __attribute__((ext_vector_type(8)))  __bf16 v8bf;
typedef __attribute__((ext_vector_type(8)))  float  v8f;

__device__ __forceinline__ v8f vzero8() {
  v8f z = {0.f, 0.f, 0.f, 0.f, 0.f, 0.f, 0.f, 0.f};
  return z;
}

// A fragment (16x32 bf16, M x K): lane holds row m=lane&15;
// elems 0..7 = K (lane>=16 ? 8 : 0)+0..7 ; elems 8..15 = K 16+(lane>=16?8:0)+0..7
__device__ __forceinline__ v16bf ld_frag_a(const bf16_t* base, int row0, int k0, int ld, int lane) {
  const bf16_t* p = base + (size_t)(row0 + (lane & 15)) * ld + k0 + ((lane >> 4) << 3);
  union { v16bf v; v8bf h[2]; } u;
  u.h[0] = *(const v8bf*)p;
  u.h[1] = *(const v8bf*)(p + 16);
  return u.v;
}

// B fragment (32x16 bf16, K x N) loaded from Bt stored N x K row-major:
// lane holds column n=lane&15; elems 0..7 = K (lane>=16 ? 8 : 0)+0..7 ; elems 8..15 = K 16+(lane>=16?8:0)+0..7
__device__ __forceinline__ v16bf ld_frag_b(const bf16_t* base, int n0, int k0, int ld, int lane) {
  const bf16_t* p = base + (size_t)(n0 + (lane & 15)) * ld + k0 + ((lane >> 4) << 3);
  union { v16bf v; v8bf h[2]; } u;
  u.h[0] = *(const v8bf*)p;
  u.h[1] = *(const v8bf*)(p + 16);
  return u.v;
}

__device__ __forceinline__ v8f wmma_bf16(v16bf a, v16bf b, v8f c) {
  return __builtin_amdgcn_wmma_f32_16x16x32_bf16(false, a, false, b, (short)0, c, false, false);
}

// ---------------- generic batched GEMM: C(MxN,f32) = A(MxK,bf16) @ Bt(NxK,bf16)^T ----------------
// flags: 1 = add bias[N], 2 = relu
// Ping-pong pipelined K loop (unrolled x2, no buffer-rotation movs): each iteration
// consumes one fragment set while the other set's global_load_b128s are in flight.
// Requires K/32 even and >= 2 (holds for all call sites: K in {128, 1024, 4096}).
// All 4 N-subtiles always computed (edge subtiles clamped to N-16) so every v_wmma
// runs with EXEC all-1s.
__global__ __launch_bounds__(256) void k_gemm(const bf16_t* __restrict__ A,
                                              const bf16_t* __restrict__ Bt,
                                              const float* __restrict__ bias,
                                              float* __restrict__ C,
                                              int M, int N, int K,
                                              long long sA, long long sB, long long sC,
                                              int flags) {
  const int lane = threadIdx.x & 31;
  const int wid  = threadIdx.x >> 5;
  const long long z = blockIdx.z;
  A += z * sA; Bt += z * sB; C += z * sC;
  const int row0 = blockIdx.y * 64  + (wid >> 1) * 16;
  const int col0 = blockIdx.x * 128 + (wid & 1) * 64;
  if (row0 >= M || col0 >= N) return;   // whole wave exits; survivors keep full EXEC

  int cb[4];
#pragma unroll
  for (int s = 0; s < 4; ++s) cb[s] = min(col0 + s * 16, N - 16);

  v8f acc[4];
#pragma unroll
  for (int s = 0; s < 4; ++s) acc[s] = vzero8();

  // prologue: fragment sets for K-steps 0 and 1
  v16bf a0 = ld_frag_a(A, row0, 0,  K, lane);
  v16bf a1 = ld_frag_a(A, row0, 32, K, lane);
  v16bf b0[4], b1[4];
#pragma unroll
  for (int s = 0; s < 4; ++s) b0[s] = ld_frag_b(Bt, cb[s], 0,  K, lane);
#pragma unroll
  for (int s = 0; s < 4; ++s) b1[s] = ld_frag_b(Bt, cb[s], 32, K, lane);

  for (int k0 = 64; k0 < K; k0 += 64) {
    if (k0 + 64 < K) {
      __builtin_prefetch(A + (size_t)(row0 + (lane & 15)) * K + k0 + 64, 0, 1);
      __builtin_prefetch(Bt + (size_t)(cb[0] + (lane & 15)) * K + k0 + 64, 0, 1);
    }
    // consume set 0, then refill it for step k0
#pragma unroll
    for (int s = 0; s < 4; ++s) acc[s] = wmma_bf16(a0, b0[s], acc[s]);
    a0 = ld_frag_a(A, row0, k0, K, lane);
#pragma unroll
    for (int s = 0; s < 4; ++s) b0[s] = ld_frag_b(Bt, cb[s], k0, K, lane);
    // consume set 1 (its loads overlap set-0 refill), then refill for step k0+32
#pragma unroll
    for (int s = 0; s < 4; ++s) acc[s] = wmma_bf16(a1, b1[s], acc[s]);
    a1 = ld_frag_a(A, row0, k0 + 32, K, lane);
#pragma unroll
    for (int s = 0; s < 4; ++s) b1[s] = ld_frag_b(Bt, cb[s], k0 + 32, K, lane);
  }
  // epilogue: consume last two sets
#pragma unroll
  for (int s = 0; s < 4; ++s) acc[s] = wmma_bf16(a0, b0[s], acc[s]);
#pragma unroll
  for (int s = 0; s < 4; ++s) acc[s] = wmma_bf16(a1, b1[s], acc[s]);

  const int n_l = lane & 15, mg = (lane >> 4) * 8;
#pragma unroll
  for (int s = 0; s < 4; ++s) {
    if (col0 + s * 16 >= N) continue;   // clamped duplicate tile -> no store
    const int col = col0 + s * 16 + n_l;
    const float bvv = (flags & 1) ? bias[col] : 0.f;
    float* cp = C + (size_t)(row0 + mg) * N + col;
#pragma unroll
    for (int r = 0; r < 8; ++r) {
      float v = acc[s][r] + bvv;
      if (flags & 2) v = fmaxf(v, 0.f);
      cp[(size_t)r * N] = v;
    }
  }
}

// ---------------- temporal conv k=3 'SAME' as 3 accumulated WMMA GEMMs ----------------
// xpad: bf16 [B][T+2][D] (zero rows at t=0 and t=T+1); w3: 3 x (D x D) bf16 (o-major, i = K)
// Flattened 96-step (tap, k0) loop, ping-pong pipelined like k_gemm (96 is even).
__global__ __launch_bounds__(256) void k_conv(const bf16_t* __restrict__ xpad,
                                              const bf16_t* __restrict__ w3,
                                              const float* __restrict__ cbv,
                                              float* __restrict__ y) {
  const int lane = threadIdx.x & 31, wid = threadIdx.x >> 5;
  const int b    = blockIdx.z;
  const int t0   = blockIdx.y * 64  + (wid >> 1) * 16;
  const int col0 = blockIdx.x * 128 + (wid & 1) * 64;
  const bf16_t* ab = xpad + (size_t)b * (Tc + 2) * Dc;
  const int n_l = lane & 15, mg = (lane >> 4) * 8;
  constexpr int KSTEPS = 3 * (Dc / 32);   // 96

  auto conv_a = [&](int kk) {
    const int k = kk >> 5, k0 = (kk & 31) << 5;     // tap, K offset (row t uses xpad row t+k)
    return ld_frag_a(ab, t0 + k, k0, Dc, lane);
  };
  auto conv_b = [&](int kk, int s) {
    const int k = kk >> 5, k0 = (kk & 31) << 5;
    return ld_frag_b(w3 + (size_t)k * Dc * Dc, col0 + s * 16, k0, Dc, lane);
  };

  v8f acc[4];
#pragma unroll
  for (int s = 0; s < 4; ++s) {
    const float bvv = cbv[col0 + s * 16 + n_l];
    v8f t;
#pragma unroll
    for (int r = 0; r < 8; ++r) t[r] = bvv;
    acc[s] = t;
  }

  v16bf a0 = conv_a(0), a1 = conv_a(1);
  v16bf b0[4], b1[4];
#pragma unroll
  for (int s = 0; s < 4; ++s) b0[s] = conv_b(0, s);
#pragma unroll
  for (int s = 0; s < 4; ++s) b1[s] = conv_b(1, s);

  for (int kk = 2; kk < KSTEPS; kk += 2) {
#pragma unroll
    for (int s = 0; s < 4; ++s) acc[s] = wmma_bf16(a0, b0[s], acc[s]);
    a0 = conv_a(kk);
#pragma unroll
    for (int s = 0; s < 4; ++s) b0[s] = conv_b(kk, s);
#pragma unroll
    for (int s = 0; s < 4; ++s) acc[s] = wmma_bf16(a1, b1[s], acc[s]);
    a1 = conv_a(kk + 1);
#pragma unroll
    for (int s = 0; s < 4; ++s) b1[s] = conv_b(kk + 1, s);
  }
#pragma unroll
  for (int s = 0; s < 4; ++s) acc[s] = wmma_bf16(a0, b0[s], acc[s]);
#pragma unroll
  for (int s = 0; s < 4; ++s) acc[s] = wmma_bf16(a1, b1[s], acc[s]);

#pragma unroll
  for (int s = 0; s < 4; ++s) {
    float* cp = y + ((size_t)(b * Tc + t0 + mg)) * Dc + col0 + s * 16 + n_l;
#pragma unroll
    for (int r = 0; r < 8; ++r) cp[(size_t)r * Dc] = acc[s][r];
  }
}

// ---------------- flash attention with Shaw RPE bias ----------------
// qh,kh: bf16 [B][H][T][DH]; vt: bf16 [B][H][DH][T]; qr: f32 [B][H][T][RP]
// o: f32 [B*T][D]  (o[b*T+i][h*DH+dh])
__global__ __launch_bounds__(128) void k_attn(const bf16_t* __restrict__ qh,
                                              const bf16_t* __restrict__ kh,
                                              const bf16_t* __restrict__ vt,
                                              const float* __restrict__ qr,
                                              float* __restrict__ o) {
  const int lane = threadIdx.x & 31;
  const int wid  = threadIdx.x >> 5;
  const int bh   = blockIdx.y;
  const int b = bh / Hc, h = bh % Hc;
  const int i0 = (blockIdx.x * 4 + wid) * 16;
  const bf16_t* qbase = qh + (size_t)bh * Tc * DHc;
  const bf16_t* kbase = kh + (size_t)bh * Tc * DHc;
  const bf16_t* vbase = vt + (size_t)bh * DHc * Tc;
  const float*  qrb   = qr + (size_t)bh * Tc * RPc;

  __shared__ float sb[4][16][34];           // per-wave P staging (C-layout -> A-layout)
  float (*pbuf)[34] = sb[wid];

  v16bf qf[4];
#pragma unroll
  for (int c = 0; c < 4; ++c) qf[c] = ld_frag_a(qbase, i0, c * 32, DHc, lane);

  v8f oacc[8];
#pragma unroll
  for (int f = 0; f < 8; ++f) oacc[f] = vzero8();
  float mrow[8], lrow[8];
#pragma unroll
  for (int r = 0; r < 8; ++r) { mrow[r] = -1e30f; lrow[r] = 0.f; }

  const int n_l = lane & 15, mg = (lane >> 4) * 8;

  for (int j0 = 0; j0 < Tc; j0 += 32) {
    // prefetch next K-tile rows and this tile's V rows; the softmax VALU section
    // below covers the V latency, the next S step covers the K latency.
    if (j0 + 32 < Tc) {
      __builtin_prefetch(kbase + (size_t)(j0 + 32 + (lane & 15)) * DHc, 0, 1);
      __builtin_prefetch(kbase + (size_t)(j0 + 48 + (lane & 15)) * DHc, 0, 1);
    }
#pragma unroll
    for (int f = 0; f < 8; f += 2)
      __builtin_prefetch(vbase + (size_t)(f * 16 + (lane & 15)) * Tc + j0, 0, 1);

    // S = Q K^T for two 16-col tiles
    v8f s0 = vzero8(), s1 = vzero8();
#pragma unroll
    for (int c = 0; c < 4; ++c) {
      v16bf k0f = ld_frag_b(kbase, j0,      c * 32, DHc, lane);
      v16bf k1f = ld_frag_b(kbase, j0 + 16, c * 32, DHc, lane);
      s0 = wmma_bf16(qf[c], k0f, s0);
      s1 = wmma_bf16(qf[c], k1f, s1);
    }
    // bias + scale + online softmax
    float p0[8], p1[8], fac[8];
#pragma unroll
    for (int r = 0; r < 8; ++r) {
      const int i = i0 + mg + r;
      int d0 = j0 + n_l - i;      d0 = d0 < -MAXOFFc ? -MAXOFFc : (d0 > MAXOFFc ? MAXOFFc : d0);
      int d1 = j0 + 16 + n_l - i; d1 = d1 < -MAXOFFc ? -MAXOFFc : (d1 > MAXOFFc ? MAXOFFc : d1);
      float v0 = (s0[r] + qrb[(size_t)i * RPc + d0 + MAXOFFc]) * SCALEc;
      float v1 = (s1[r] + qrb[(size_t)i * RPc + d1 + MAXOFFc]) * SCALEc;
      float tm = fmaxf(v0, v1);
#pragma unroll
      for (int m = 1; m < 16; m <<= 1) tm = fmaxf(tm, __shfl_xor(tm, m, 32));
      const float nm = fmaxf(mrow[r], tm);
      const float fc = __expf(mrow[r] - nm);
      const float e0 = __expf(v0 - nm);
      const float e1 = __expf(v1 - nm);
      float rsum = e0 + e1;
#pragma unroll
      for (int m = 1; m < 16; m <<= 1) rsum += __shfl_xor(rsum, m, 32);
      lrow[r] = lrow[r] * fc + rsum;
      mrow[r] = nm;
      fac[r] = fc; p0[r] = e0; p1[r] = e1;
    }
#pragma unroll
    for (int f = 0; f < 8; ++f)
#pragma unroll
      for (int r = 0; r < 8; ++r) oacc[f][r] *= fac[r];

    // stage P (C-layout) into LDS, read back as A fragment (16x32) in bf16
#pragma unroll
    for (int r = 0; r < 8; ++r) {
      pbuf[mg + r][n_l]      = p0[r];
      pbuf[mg + r][16 + n_l] = p1[r];
    }
    asm volatile("s_wait_dscnt 0" ::: "memory");
    v16bf pf;
    {
      const int m_l = lane & 15, kq = (lane >> 4) * 8;
#pragma unroll
      for (int i2 = 0; i2 < 8; ++i2) {
        pf[i2]     = (bf16_t)pbuf[m_l][kq + i2];
        pf[8 + i2] = (bf16_t)pbuf[m_l][16 + kq + i2];
      }
    }
    // O += P @ V  (Vt rows are dh, K = j)
#pragma unroll
    for (int f = 0; f < 8; ++f) {
      v16bf vf = ld_frag_b(vbase, f * 16, j0, Tc, lane);
      oacc[f] = wmma_bf16(pf, vf, oacc[f]);
    }
  }

  float inv[8];
#pragma unroll
  for (int r = 0; r < 8; ++r) inv[r] = 1.f / lrow[r];
#pragma unroll
  for (int f = 0; f < 8; ++f) {
    const int dh = f * 16 + n_l;
#pragma unroll
    for (int r = 0; r < 8; ++r) {
      const int i = i0 + mg + r;
      o[((size_t)(b * Tc + i)) * Dc + h * DHc + dh] = oacc[f][r] * inv[r];
    }
  }
}

// ---------------- elementwise / reshape helpers ----------------
__global__ __launch_bounds__(256) void k_embed(const float* __restrict__ oh,
                                               const float* __restrict__ pw,
                                               const float* __restrict__ pb,
                                               float* __restrict__ x) {
  const size_t idx = (size_t)blockIdx.x * 256 + threadIdx.x;
  if (idx >= (size_t)Mtok * Dc) return;
  const int d = (int)(idx % Dc);
  const size_t m = idx / Dc;
  float a = pb[d];
  const float* op = oh + m * NCc;
#pragma unroll 8
  for (int c = 0; c < NCc; ++c) a += op[c] * pw[(size_t)c * Dc + d];
  x[idx] = a;
}

__global__ __launch_bounds__(256) void k_pad(const float* __restrict__ x, bf16_t* __restrict__ xpad) {
  const size_t idx = (size_t)blockIdx.x * 256 + threadIdx.x;
  if (idx >= (size_t)Bc * (Tc + 2) * Dc) return;
  const int d  = (int)(idx % Dc);
  const int tt = (int)((idx / Dc) % (Tc + 2));
  const int b  = (int)(idx / ((size_t)Dc * (Tc + 2)));
  float v = 0.f;
  if (tt > 0 && tt < Tc + 1) v = x[((size_t)b * Tc + tt - 1) * Dc + d];
  xpad[idx] = (bf16_t)v;
}

// dst[n*K + k] = (n < Nvalid) ? src[k*rs + n*cs] : 0   (transpose+convert to bf16 NxK)
__global__ __launch_bounds__(256) void k_cvt(const float* __restrict__ src, bf16_t* __restrict__ dst,
                                             int Kd, int Nd, long long rs, long long cs, int Nvalid) {
  const size_t idx = (size_t)blockIdx.x * 256 + threadIdx.x;
  if (idx >= (size_t)Nd * Kd) return;
  const int n = (int)(idx / Kd);
  const int k = (int)(idx % Kd);
  float v = 0.f;
  if (n < Nvalid) v = src[(size_t)k * rs + (size_t)n * cs];
  dst[idx] = (bf16_t)v;
}

__global__ __launch_bounds__(256) void k_f2bf(const float* __restrict__ s, bf16_t* __restrict__ d, size_t n) {
  const size_t idx = (size_t)blockIdx.x * 256 + threadIdx.x;
  if (idx < n) d[idx] = (bf16_t)s[idx];
}

// mode 0: (M,D) f32 -> [B][H][T][DH] bf16 ; mode 1: -> [B][H][DH][T] bf16
__global__ __launch_bounds__(256) void k_split(const float* __restrict__ src, bf16_t* __restrict__ dst, int mode) {
  const size_t idx = (size_t)blockIdx.x * 256 + threadIdx.x;
  if (idx >= (size_t)Mtok * Dc) return;
  const size_t m = idx / Dc;
  const int col = (int)(idx % Dc);
  const int b = (int)(m / Tc), t = (int)(m % Tc);
  const int h = col / DHc, dh = col % DHc;
  const size_t off = (mode == 0)
      ? ((((size_t)b * Hc + h) * Tc + t) * DHc + dh)
      : ((((size_t)b * Hc + h) * DHc + dh) * Tc + t);
  dst[off] = (bf16_t)src[idx];
}

// out[row] = LN(addres ? xin+yin : xin) * g + bt ; population variance, eps 1e-5
__global__ __launch_bounds__(256) void k_layernorm(float* __restrict__ out,
                                                   const float* __restrict__ xin,
                                                   const float* __restrict__ yin,
                                                   const float* __restrict__ g,
                                                   const float* __restrict__ bt,
                                                   int addres) {
  const int row = blockIdx.x, tid = threadIdx.x;
  float v[4]; float s = 0.f, q = 0.f;
#pragma unroll
  for (int e = 0; e < 4; ++e) {
    const int d = tid + e * 256;
    float a = xin[(size_t)row * Dc + d];
    if (addres) a += yin[(size_t)row * Dc + d];
    v[e] = a; s += a; q += a * a;
  }
  __shared__ float rs[256], rq[256];
  rs[tid] = s; rq[tid] = q; __syncthreads();
  for (int off = 128; off > 0; off >>= 1) {
    if (tid < off) { rs[tid] += rs[tid + off]; rq[tid] += rq[tid + off]; }
    __syncthreads();
  }
  const float mean = rs[0] * (1.f / Dc);
  const float var  = rq[0] * (1.f / Dc) - mean * mean;
  const float rstd = rsqrtf(var + 1e-5f);
#pragma unroll
  for (int e = 0; e < 4; ++e) {
    const int d = tid + e * 256;
    out[(size_t)row * Dc + d] = (v[e] - mean) * rstd * g[d] + bt[d];
  }
}

// ---------------- host orchestration ----------------
extern "C" void kernel_launch(void* const* d_in, const int* in_sizes, int n_in,
                              void* d_out, int out_size, void* d_ws, size_t ws_size,
                              hipStream_t stream) {
  (void)in_sizes; (void)n_in; (void)out_size; (void)ws_size;
  const float* gt      = (const float*)d_in[0];
  const float* proj_w  = (const float*)d_in[1];
  const float* proj_b  = (const float*)d_in[2];
  const float* conv_w  = (const float*)d_in[3];
  const float* conv_b  = (const float*)d_in[4];
  const float* rpe     = (const float*)d_in[5];
  const float* wq      = (const float*)d_in[6];
  const float* wk      = (const float*)d_in[7];
  const float* wv      = (const float*)d_in[8];
  const float* wo      = (const float*)d_in[9];
  const float* bq      = (const float*)d_in[10];
  const float* bk      = (const float*)d_in[11];
  const float* bv      = (const float*)d_in[12];
  const float* bo      = (const float*)d_in[13];
  const float* ln1s    = (const float*)d_in[14];
  const float* ln1b    = (const float*)d_in[15];
  const float* ln2s    = (const float*)d_in[16];
  const float* ln2b    = (const float*)d_in[17];
  const float* w1      = (const float*)d_in[18];
  const float* b1      = (const float*)d_in[19];
  const float* w2      = (const float*)d_in[20];
  const float* b2      = (const float*)d_in[21];
  const float* normfs  = (const float*)d_in[22];
  const float* normfb  = (const float*)d_in[23];
  const float* finew   = (const float*)d_in[24];
  const float* fineb   = (const float*)d_in[25];
  const float* coarsew = (const float*)d_in[26];
  const float* coarseb = (const float*)d_in[27];
  float* outf = (float*)d_out;

  // workspace carve (256B aligned slabs, ~321 MB total)
  char* p = (char*)d_ws;
  auto take = [&](size_t n) -> char* { char* r = p; p += (n + 255) & ~(size_t)255; return r; };
  float*  x      = (float*) take((size_t)Mtok * Dc * 4);
  bf16_t* xbf    = (bf16_t*)take((size_t)Mtok * Dc * 2);
  bf16_t* xpad   = (bf16_t*)take((size_t)Bc * (Tc + 2) * Dc * 2);
  bf16_t* wt     = (bf16_t*)take((size_t)Dc * DFFc * 2);       // reused weight buffer (holds 3 conv taps too)
  bf16_t* rpebf  = (bf16_t*)take((size_t)RPc * DHc * 2);
  bf16_t* qhb    = (bf16_t*)take((size_t)Mtok * Dc * 2);
  bf16_t* khb    = (bf16_t*)take((size_t)Mtok * Dc * 2);
  bf16_t* vtb    = (bf16_t*)take((size_t)Mtok * Dc * 2);
  float*  qrbuf  = (float*) take((size_t)Bc * Hc * Tc * RPc * 4);
  float*  fbig   = (float*) take((size_t)Mtok * DFFc * 4);
  bf16_t* fbigbf = (bf16_t*)take((size_t)Mtok * DFFc * 2);
  float*  c1     = (float*) take((size_t)Mtok * Dc * 4);

  const dim3 blk256(256), blk128(128);
  auto ew = [](size_t n) { return dim3((unsigned)((n + 255) / 256)); };
  auto gemm = [&](const bf16_t* A, const bf16_t* Bt, const float* bias, float* C,
                  int Mm, int Nn, int Kk, int flags, int batch,
                  long long sA, long long sB, long long sC) {
    k_gemm<<<dim3((Nn + 127) / 128, (Mm + 63) / 64, batch), dim3(256), 0, stream>>>(
        A, Bt, bias, C, Mm, Nn, Kk, sA, sB, sC, flags);
  };

  // embed + temporal conv
  k_embed<<<ew((size_t)Mtok * Dc), blk256, 0, stream>>>(gt, proj_w, proj_b, x);
  k_pad<<<ew((size_t)Bc * (Tc + 2) * Dc), blk256, 0, stream>>>(x, xpad);
  for (int k = 0; k < 3; ++k)
    k_cvt<<<ew((size_t)Dc * Dc), blk256, 0, stream>>>(conv_w + k, wt + (size_t)k * Dc * Dc,
                                                      Dc, Dc, 3, 3LL * Dc, Dc);
  k_conv<<<dim3(Dc / 128, Tc / 64, Bc), blk256, 0, stream>>>(xpad, wt, conv_b, x);
  // rpe table -> bf16 (already N x K with N=r, K=dh), zero-padded to RP rows
  k_cvt<<<ew((size_t)RPc * DHc), blk256, 0, stream>>>(rpe, rpebf, DHc, RPc, 1, DHc, Rc);

  for (int l = 0; l < Lc; ++l) {
    k_f2bf<<<ew((size_t)Mtok * Dc), blk256, 0, stream>>>(x, xbf, (size_t)Mtok * Dc);
    // Q / K / V projections (reuse fbig as fp32 staging, split into head layouts)
    k_cvt<<<ew((size_t)Dc * Dc), blk256, 0, stream>>>(wq + (size_t)l * Dc * Dc, wt, Dc, Dc, Dc, 1, Dc);
    gemm(xbf, wt, bq + (size_t)l * Dc, fbig, Mtok, Dc, Dc, 1, 1, 0, 0, 0);
    k_split<<<ew((size_t)Mtok * Dc), blk256, 0, stream>>>(fbig, qhb, 0);
    k_cvt<<<ew((size_t)Dc * Dc), blk256, 0, stream>>>(wk + (size_t)l * Dc * Dc, wt, Dc, Dc, Dc, 1, Dc);
    gemm(xbf, wt, bk + (size_t)l * Dc, fbig, Mtok, Dc, Dc, 1, 1, 0, 0, 0);
    k_split<<<ew((size_t)Mtok * Dc), blk256, 0, stream>>>(fbig, khb, 0);
    k_cvt<<<ew((size_t)Dc * Dc), blk256, 0, stream>>>(wv + (size_t)l * Dc * Dc, wt, Dc, Dc, Dc, 1, Dc);
    gemm(xbf, wt, bv + (size_t)l * Dc, fbig, Mtok, Dc, Dc, 1, 1, 0, 0, 0);
    k_split<<<ew((size_t)Mtok * Dc), blk256, 0, stream>>>(fbig, vtb, 1);
    // qr[b,h] = q @ rpe^T  (batched over B*H, shared Bt)
    gemm(qhb, rpebf, nullptr, qrbuf, Tc, RPc, DHc, 0, Bc * Hc,
         (long long)Tc * DHc, 0, (long long)Tc * RPc);
    // flash attention with RPE bias
    k_attn<<<dim3(Tc / 64, Bc * Hc), blk128, 0, stream>>>(qhb, khb, vtb, qrbuf, c1);
    // output projection + residual + LN1
    k_f2bf<<<ew((size_t)Mtok * Dc), blk256, 0, stream>>>(c1, xbf, (size_t)Mtok * Dc);
    k_cvt<<<ew((size_t)Dc * Dc), blk256, 0, stream>>>(wo + (size_t)l * Dc * Dc, wt, Dc, Dc, Dc, 1, Dc);
    gemm(xbf, wt, bo + (size_t)l * Dc, c1, Mtok, Dc, Dc, 1, 1, 0, 0, 0);
    k_layernorm<<<dim3(Mtok), blk256, 0, stream>>>(x, x, c1, ln1s + (size_t)l * Dc, ln1b + (size_t)l * Dc, 1);
    // MLP
    k_f2bf<<<ew((size_t)Mtok * Dc), blk256, 0, stream>>>(x, xbf, (size_t)Mtok * Dc);
    k_cvt<<<ew((size_t)Dc * DFFc), blk256, 0, stream>>>(w1 + (size_t)l * Dc * DFFc, wt, Dc, DFFc, DFFc, 1, DFFc);
    gemm(xbf, wt, b1 + (size_t)l * DFFc, fbig, Mtok, DFFc, Dc, 1 | 2, 1, 0, 0, 0);
    k_f2bf<<<ew((size_t)Mtok * DFFc), blk256, 0, stream>>>(fbig, fbigbf, (size_t)Mtok * DFFc);
    k_cvt<<<ew((size_t)DFFc * Dc), blk256, 0, stream>>>(w2 + (size_t)l * DFFc * Dc, wt, DFFc, Dc, Dc, 1, Dc);
    gemm(fbigbf, wt, b2 + (size_t)l * Dc, c1, Mtok, Dc, DFFc, 1, 1, 0, 0, 0);
    k_layernorm<<<dim3(Mtok), blk256, 0, stream>>>(x, x, c1, ln2s + (size_t)l * Dc, ln2b + (size_t)l * Dc, 1);
  }

  // final LN + output heads (directly into d_out: fine then coarse)
  k_layernorm<<<dim3(Mtok), blk256, 0, stream>>>(c1, x, x, normfs, normfb, 0);
  k_f2bf<<<ew((size_t)Mtok * Dc), blk256, 0, stream>>>(c1, xbf, (size_t)Mtok * Dc);
  k_cvt<<<ew((size_t)NCc * Dc), blk256, 0, stream>>>(finew, wt, Dc, NCc, NCc, 1, NCc);
  gemm(xbf, wt, fineb, outf, Mtok, NCc, Dc, 1, 1, 0, 0, 0);
  k_cvt<<<ew((size_t)NCc * Dc), blk256, 0, stream>>>(coarsew, wt, Dc, NCc, NCc, 1, NCc);
  gemm(xbf, wt, coarseb, outf + (size_t)Mtok * NCc, Mtok, NCc, Dc, 1, 1, 0, 0, 0);
}
